// GaussianMixUp_83073257439577
// MI455X (gfx1250) — matrix operand order repaired and checked by reading, not compiled, not run on hardware
//
#include <hip/hip_runtime.h>

typedef __attribute__((ext_vector_type(16))) __bf16 v16bf;
typedef __attribute__((ext_vector_type(8)))  float  v8f;

#define KMAX 45
#define PAD  22          // (KMAX-1)/2, SAME padding
#define NF   90
#define TOPKN 64
#define NPLANE 12        // B*C = 4*3
#define H 256
#define W 256
#define XTS  72          // transposed input stage: sXT[col][row], stride 72 (144B, 16B-aligned)
#define TTS  72          // transposed horizontal Toeplitz stride
#define YSTRIDE 320      // vertical-pass result rows, cols [-22,298) with zero halo

#define WMMA_BF16(A,B,C) \
  __builtin_amdgcn_wmma_f32_16x16x32_bf16(false,(A),false,(B),(short)0,(C),false,false)

// ---- fragment gathers (wave32 layouts per CDNA5 ISA 7.12.2) -----------------
// A (16x32 bf16): lanes 0-15 -> M=lane, elems 0..7 = K 0..7, elems 8..15 = K 16..23
//                 lanes 16-31 -> M=lane-16, elems 0..7 = K 8..15, elems 8..15 = K 24..31
// Rows are K-contiguous: two 8-element (16B) runs per lane -> ds_load_b128 x2.
__device__ __forceinline__ v16bf load_frag_a(const __bf16* base, int stride,
                                             int colOff, int kc, int lane) {
  const int m  = lane & 15;
  const int hi = lane >> 4;
  const __bf16* row = base + m * stride + colOff;
  const int b0 = kc + hi * 8;
  const int b1 = kc + 16 + hi * 8;
  v16bf f;
#pragma unroll
  for (int e = 0; e < 8; ++e) f[e] = row[b0 + e];
#pragma unroll
  for (int e = 0; e < 8; ++e) f[8 + e] = row[b1 + e];
  return f;
}

// B (32x16 bf16): lanes 0-15 -> N=lane, K 0..7,16..23 ; lanes 16-31 -> N=lane-16, K 8..15,24..31.
// Source is stored TRANSPOSED ([n][k]) so each lane reads two contiguous 8-element (16B) runs, 16 apart.
__device__ __forceinline__ v16bf load_frag_b_t(const __bf16* baseT, int stride,
                                               int colOff, int kc, int lane) {
  const int n  = lane & 15;
  const int kb = kc + (lane >> 4) * 8;
  const __bf16* p = baseT + (colOff + n) * stride + kb;
  v16bf f;
#pragma unroll
  for (int e = 0; e < 8; ++e) f[e] = p[e];
#pragma unroll
  for (int e = 0; e < 8; ++e) f[8 + e] = p[16 + e];
  return f;
}

// ---- prep: sigmoid(lambda) + stable top-64 of 90 (ties -> lower index) ------
__global__ void gmix_prep_kernel(const float* __restrict__ prompt,
                                 const float* __restrict__ lambda_param,
                                 float* __restrict__ lam_out,
                                 int* __restrict__ idx_out) {
  if (threadIdx.x == 0) {
    lam_out[0] = 1.0f / (1.0f + __expf(-lambda_param[0]));
    float v[NF];
    for (int i = 0; i < NF; ++i) v[i] = prompt[i];
    for (int t = 0; t < TOPKN; ++t) {
      int best = 0; float bv = v[0];
      for (int i = 1; i < NF; ++i) { if (v[i] > bv) { bv = v[i]; best = i; } }
      idx_out[t] = best;
      v[best] = -3.0e38f;
    }
  }
}

// ---- main: per (pair, plane, 16-row strip) ---------------------------------
__global__ __launch_bounds__(256)
void gmix_main_kernel(const float* __restrict__ img, const float* __restrict__ kern,
                      const float* __restrict__ lamp, const int* __restrict__ idx,
                      float* __restrict__ out) {
  __shared__ __bf16 sXT[W * XTS];           // input strip, TRANSPOSED: [col][row 0..63]
  __shared__ __bf16 sY[2][16 * YSTRIDE];    // vertical-pass result (row-major), both filters
  __shared__ __bf16 sAv[2][16 * 64];        // vertical Toeplitz  Av[m][r] = g[r-m]
  __shared__ __bf16 sThT[2][16 * TTS];      // horizontal Toeplitz TRANSPOSED: [n][r] = g[r-n]

  const int tid   = threadIdx.x;
  const int lane  = tid & 31;
  const int wave  = tid >> 5;
  const int strip = blockIdx.x;             // 0..15
  const int plane = blockIdx.y;             // 0..11
  const int pair  = blockIdx.z;             // 0..31
  const int h0    = strip * 16;
  const int f0 = idx[2 * pair], f1 = idx[2 * pair + 1];
  const float lam = lamp[0];
  const float* imgP = img + plane * (H * W);

  // stage rows [h0-22, h0+41] (64 rows), zero outside; coalesced read, transposed write
  for (int i = tid; i < 64 * W; i += 256) {
    int r = i >> 8, c = i & 255;
    int gr = h0 - PAD + r;
    float v = (gr >= 0 && gr < H) ? imgP[gr * W + c] : 0.0f;
    sXT[c * XTS + r] = (__bf16)v;
  }
  // banded Toeplitz operators for both selected filters
  for (int i = tid; i < 2 * 16 * 64; i += 256) {
    int f = i >> 10, m = (i >> 6) & 15, r = i & 63;
    int k = r - m;
    float v = (k >= 0 && k < KMAX) ? kern[(f ? f1 : f0) * KMAX + k] : 0.0f;
    sAv[f][m * 64 + r] = (__bf16)v;
  }
  for (int i = tid; i < 2 * 16 * 64; i += 256) {
    int f = i >> 10, n = (i >> 6) & 15, r = i & 63;
    int k = r - n;
    float v = (k >= 0 && k < KMAX) ? kern[(f ? f1 : f0) * KMAX + k] : 0.0f;
    sThT[f][n * TTS + r] = (__bf16)v;
  }
  // zero Y (covers the +/-22 column halo); 32-bit writes
  for (int i = tid; i < 2 * 16 * YSTRIDE / 2; i += 256)
    ((unsigned int*)sY)[i] = 0u;

  // warm L2 with the next strip's rows (speculative prefetch)
  if (strip < 15 && tid < 64)
    __builtin_prefetch(imgP + (h0 + 16 + (tid >> 2)) * W + (tid & 3) * 64, 0, 0);

  __syncthreads();

  // Phase 1: vertical pass. Y[f] tile = Av[f](16x64) x Xext(64x16), K split 0/32.
#pragma unroll
  for (int it = 0; it < 4; ++it) {
    const int jb = wave + 8 * it;           // 32 jobs: filter x 16 col-tiles
    const int f  = jb >> 4;
    const int w0 = (jb & 15) * 16;
    v8f acc = {0.f,0.f,0.f,0.f,0.f,0.f,0.f,0.f};
    v16bf a0 = load_frag_a(sAv[f], 64, 0, 0, lane);
    v16bf b0 = load_frag_b_t(sXT, XTS, w0, 0, lane);
    acc = WMMA_BF16(a0, b0, acc);
    v16bf a1 = load_frag_a(sAv[f], 64, 0, 32, lane);
    v16bf b1 = load_frag_b_t(sXT, XTS, w0, 32, lane);
    acc = WMMA_BF16(a1, b1, acc);
    const int n = lane & 15, hi = lane >> 4;
#pragma unroll
    for (int i = 0; i < 8; ++i) {
      int m = i + hi * 8;
      sY[f][m * YSTRIDE + PAD + w0 + n] = (__bf16)acc[i];
    }
  }
  __syncthreads();

  // Phase 2: horizontal pass + clamp + lambda mix.
  float* outP = out + (pair * NPLANE + plane) * (H * W);
#pragma unroll
  for (int it = 0; it < 2; ++it) {
    const int j  = wave + 8 * it;           // 16 col-tiles
    const int w0 = j * 16;
    v8f acc0 = {0.f,0.f,0.f,0.f,0.f,0.f,0.f,0.f};
    v8f acc1 = {0.f,0.f,0.f,0.f,0.f,0.f,0.f,0.f};
    {
      v16bf a = load_frag_a(sY[0], YSTRIDE, w0, 0, lane);
      v16bf b = load_frag_b_t(sThT[0], TTS, 0, 0, lane);
      acc0 = WMMA_BF16(a, b, acc0);
      a = load_frag_a(sY[0], YSTRIDE, w0, 32, lane);
      b = load_frag_b_t(sThT[0], TTS, 0, 32, lane);
      acc0 = WMMA_BF16(a, b, acc0);
    }
    {
      v16bf a = load_frag_a(sY[1], YSTRIDE, w0, 0, lane);
      v16bf b = load_frag_b_t(sThT[1], TTS, 0, 0, lane);
      acc1 = WMMA_BF16(a, b, acc1);
      a = load_frag_a(sY[1], YSTRIDE, w0, 32, lane);
      b = load_frag_b_t(sThT[1], TTS, 0, 32, lane);
      acc1 = WMMA_BF16(a, b, acc1);
    }
    const int n = lane & 15, hi = lane >> 4;
#pragma unroll
    for (int i = 0; i < 8; ++i) {
      int m = i + hi * 8;
      int row = h0 + m, col = w0 + n;
      float iv = imgP[row * W + col];
      float hf0 = fminf(fmaxf(iv - acc0[i], -1.0f), 1.0f);
      float hf1 = fminf(fmaxf(iv - acc1[i], -1.0f), 1.0f);
      outP[row * W + col] = lam * hf0 + (1.0f - lam) * hf1;
    }
  }
}

extern "C" void kernel_launch(void* const* d_in, const int* in_sizes, int n_in,
                              void* d_out, int out_size, void* d_ws, size_t ws_size,
                              hipStream_t stream) {
  const float* img          = (const float*)d_in[0];   // [4,3,256,256]
  const float* kern         = (const float*)d_in[1];   // [90,45]
  const float* prompt       = (const float*)d_in[2];   // [90]
  const float* lambda_param = (const float*)d_in[3];   // [1]

  float* lam = (float*)d_ws;
  int*   idx = (int*)((char*)d_ws + 16);

  gmix_prep_kernel<<<1, 64, 0, stream>>>(prompt, lambda_param, lam, idx);

  dim3 grid(16, NPLANE, 32);   // strips x planes x pairs
  gmix_main_kernel<<<grid, 256, 0, stream>>>(img, kern, lam, idx, (float*)d_out);
}
